// TransformerEncoder_1889785610583
// MI455X (gfx1250) — matrix operand order repaired and checked by reading, not compiled, not run on hardware
//
#include <hip/hip_runtime.h>
#include <hip/hip_bf16.h>

// ---- problem constants (match reference) ----
constexpr int L_ = 6, D_ = 512, FF_ = 2048, H_ = 8, S_ = 1024, B_ = 4, DH_ = 64;
constexpr int NTOK_ = B_ * S_;          // 4096 tokens
constexpr float EPS_ = 1e-5f;
constexpr float SCALE_ = 0.125f;        // 1/sqrt(DH)

typedef __attribute__((ext_vector_type(16))) _Float16 v16h;
typedef __attribute__((ext_vector_type(8)))  _Float16 v8h;
typedef __attribute__((ext_vector_type(8)))  float    v8f;

union U16h { v16h v; v8h h[2]; };
union U8h  { v8h v; _Float16 e[8]; };

// ------------------------------------------------------------------
// Register-blocked WMMA: each wave computes MT x NT 16x16 f32 tiles.
// MT=4, NT=2 (64x32) keeps total VGPR demand ~112 + pipelining head-
// room -> no scratch spills (4x4 spilled in round 2).
// A (row-major, lda), Bt (= B^T, row-major N x K, ldb).
// ------------------------------------------------------------------
template <int MT, int NT>
__device__ inline void wmma_block(const _Float16* __restrict__ A, int lda,
                                  const _Float16* __restrict__ Bt, int ldb,
                                  int K, int lane, v8f c[MT][NT]) {
  const int r  = lane & 15;
  const int hi = lane >> 4;
  const _Float16* ap = A  + (size_t)r * lda + hi * 8;
  const _Float16* bp = Bt + (size_t)r * ldb + hi * 8;
  for (int k0 = 0; k0 < K; k0 += 32) {
    U16h a[MT], b[NT];
#pragma unroll
    for (int i = 0; i < MT; i++) {
      const _Float16* q = ap + (size_t)i * 16 * lda + k0;
      a[i].h[0] = *(const v8h*)q;
      a[i].h[1] = *(const v8h*)(q + 16);
    }
#pragma unroll
    for (int j = 0; j < NT; j++) {
      const _Float16* q = bp + (size_t)j * 16 * ldb + k0;
      b[j].h[0] = *(const v8h*)q;
      b[j].h[1] = *(const v8h*)(q + 16);
    }
#pragma unroll
    for (int i = 0; i < MT; i++)
#pragma unroll
      for (int j = 0; j < NT; j++)
        c[i][j] = __builtin_amdgcn_wmma_f32_16x16x32_f16(
            false, a[i].v, false, b[j].v, (short)0, c[i][j], false, false);
  }
}

// ---- block reductions (256 threads) ----
__device__ inline float block_sum(float v, float* sbuf) {
  int t = threadIdx.x;
  sbuf[t] = v; __syncthreads();
  for (int s = 128; s > 0; s >>= 1) {
    if (t < s) sbuf[t] += sbuf[t + s];
    __syncthreads();
  }
  float r = sbuf[0]; __syncthreads();
  return r;
}
__device__ inline float block_max(float v, float* sbuf) {
  int t = threadIdx.x;
  sbuf[t] = v; __syncthreads();
  for (int s = 128; s > 0; s >>= 1) {
    if (t < s) sbuf[t] = fmaxf(sbuf[t], sbuf[t + s]);
    __syncthreads();
  }
  float r = sbuf[0]; __syncthreads();
  return r;
}

// ------------------------------------------------------------------
// Weight convert + transpose: W (K x N, f32) -> Wt (N x K, f16)
// ------------------------------------------------------------------
__global__ void k_cvt_wt(const float* __restrict__ W, _Float16* __restrict__ Wt,
                         int K, int N) {
  int i = blockIdx.x * 256 + threadIdx.x;   // over N*K, output-indexed
  int n = i / K, k = i % K;
  Wt[i] = (_Float16)W[(size_t)k * N + n];
}

// x(b,s,d) = src(s,b,d) + pos(s,b,d)
__global__ void k_add_pos(const float* __restrict__ src,
                          const float* __restrict__ pos,
                          float* __restrict__ x) {
  int i = blockIdx.x * 256 + threadIdx.x;   // flat over (S,B,D)
  int d = i % D_;
  int b = (i / D_) % B_;
  int s = i / (D_ * B_);
  x[((size_t)b * S_ + s) * D_ + d] = src[i] + pos[i];
}

// LayerNorm over D=512, f16 output (for GEMM A operand). 256 thr, 2 elem/thr.
__global__ void k_ln(const float* __restrict__ x, const float* __restrict__ g,
                     const float* __restrict__ bb, _Float16* __restrict__ out) {
  __shared__ float sbuf[256];
  int tok = blockIdx.x, t = threadIdx.x;
  const float* xr = x + (size_t)tok * D_;
  float v0 = xr[t], v1 = xr[t + 256];
  float m = block_sum(v0 + v1, sbuf) * (1.f / D_);
  float d0 = v0 - m, d1 = v1 - m;
  float var = block_sum(d0 * d0 + d1 * d1, sbuf) * (1.f / D_);
  float inv = rsqrtf(var + EPS_);
  _Float16* o = out + (size_t)tok * D_;
  o[t]       = (_Float16)(d0 * inv * g[t] + bb[t]);
  o[t + 256] = (_Float16)(d1 * inv * g[t + 256] + bb[t + 256]);
}

// Final LayerNorm: f32 out, transposed back to (S,B,D)
__global__ void k_final_ln(const float* __restrict__ x, const float* __restrict__ g,
                           const float* __restrict__ bb, float* __restrict__ out) {
  __shared__ float sbuf[256];
  int tok = blockIdx.x, t = threadIdx.x;
  int b = tok / S_, s = tok % S_;
  const float* xr = x + (size_t)tok * D_;
  float v0 = xr[t], v1 = xr[t + 256];
  float m = block_sum(v0 + v1, sbuf) * (1.f / D_);
  float d0 = v0 - m, d1 = v1 - m;
  float var = block_sum(d0 * d0 + d1 * d1, sbuf) * (1.f / D_);
  float inv = rsqrtf(var + EPS_);
  float* o = out + ((size_t)s * B_ + b) * D_;
  o[t]       = d0 * inv * g[t] + bb[t];
  o[t + 256] = d1 * inv * g[t + 256] + bb[t + 256];
}

// Softmax over rows of S=1024, in-place on f16. One block (256 thr) per row.
__global__ void k_softmax(_Float16* __restrict__ att) {
  __shared__ float sbuf[256];
  size_t row = blockIdx.x;
  _Float16* pr = att + row * S_;
  int t = threadIdx.x;
  float v[4];
  float mx = -1e30f;
#pragma unroll
  for (int i = 0; i < 4; i++) { v[i] = (float)pr[t + i * 256]; mx = fmaxf(mx, v[i]); }
  mx = block_max(mx, sbuf);
  float sum = 0.f;
#pragma unroll
  for (int i = 0; i < 4; i++) { v[i] = __expf(v[i] - mx); sum += v[i]; }
  sum = block_sum(sum, sbuf);
  float inv = 1.f / sum;
#pragma unroll
  for (int i = 0; i < 4; i++) pr[t + i * 256] = (_Float16)(v[i] * inv);
}

// ------------------------------------------------------------------
// GEMM: out16(4096 x N) = A(4096 x K,f16) * Bt^T + bias, optional ReLU.
// Wave tile 64x32 (4x2 wmma tiles). Block = dim3(32,8) = 8 waves.
// Q/K: N=K=512 (grid 128).  FFN1: N=2048,K=512,ReLU (grid 512).
// ------------------------------------------------------------------
template <bool RELU>
__global__ void __launch_bounds__(256, 1)
k_gemm_bias16(const _Float16* __restrict__ A,
              const _Float16* __restrict__ Bt,
              const float* __restrict__ bias,
              _Float16* __restrict__ out, int N, int K) {
  const int lane = threadIdx.x;
  const int w = blockIdx.x * 8 + threadIdx.y;
  const int nwn = N >> 5;                    // N / 32
  const int wm = w / nwn, wn = w % nwn;
  v8f zero = {};
  v8f c[4][2];
#pragma unroll
  for (int i = 0; i < 4; i++)
#pragma unroll
    for (int j = 0; j < 2; j++) c[i][j] = zero;
  wmma_block<4, 2>(A + (size_t)wm * 64 * K, K, Bt + (size_t)wn * 32 * K, K, K, lane, c);
  const int hi = lane >> 4;
#pragma unroll
  for (int j = 0; j < 2; j++) {
    const int nn = wn * 32 + j * 16 + (lane & 15);
    const float bv = bias[nn];
#pragma unroll
    for (int i = 0; i < 4; i++) {
      _Float16* orow = out + (size_t)(wm * 64 + i * 16 + hi * 8) * N + nn;
#pragma unroll
      for (int r = 0; r < 8; r++) {
        float vv = c[i][j][r] + bv;
        if (RELU) vv = fmaxf(vv, 0.f);
        orow[(size_t)r * N] = (_Float16)vv;
      }
    }
  }
}

// V projection with transposed store: vT[b][h][dh][s] (contiguous in s).
// N=K=D=512, wave tile 64x32, grid 128.
__global__ void __launch_bounds__(256, 1)
k_gemm_v(const _Float16* __restrict__ A,
         const _Float16* __restrict__ Bt,
         const float* __restrict__ bias,
         _Float16* __restrict__ vT) {
  const int lane = threadIdx.x;
  const int w = blockIdx.x * 8 + threadIdx.y;
  const int wm = w >> 4, wn = w & 15;        // nwn = 512/32 = 16
  v8f zero = {};
  v8f c[4][2];
#pragma unroll
  for (int i = 0; i < 4; i++)
#pragma unroll
    for (int j = 0; j < 2; j++) c[i][j] = zero;
  wmma_block<4, 2>(A + (size_t)wm * 64 * D_, D_, Bt + (size_t)wn * 32 * D_, D_, D_, lane, c);
  const int hi = lane >> 4;
#pragma unroll
  for (int j = 0; j < 2; j++) {
    const int nn = wn * 32 + j * 16 + (lane & 15);
    const int h = nn >> 6, dh = nn & 63;     // DH = 64
    const float bv = bias[nn];
#pragma unroll
    for (int i = 0; i < 4; i++) {
      const int m0 = wm * 64 + i * 16;
      const int b = m0 >> 10;                // S = 1024
      const int s0 = (m0 & 1023) + hi * 8;
      U8h u;
#pragma unroll
      for (int r = 0; r < 8; r++) u.e[r] = (_Float16)(c[i][j][r] + bv);
      *(v8h*)(vT + (((size_t)(b * H_ + h) * DH_) + dh) * S_ + s0) = u.v;
    }
  }
}

// scores = scale * Q Kt : per (b,h) S x S. Wave tile 64x32, grid (64, B*H).
__global__ void __launch_bounds__(256, 1)
k_gemm_scores(const _Float16* __restrict__ q,
              const _Float16* __restrict__ k,
              _Float16* __restrict__ att) {
  const int lane = threadIdx.x;
  const int bh = blockIdx.y;
  const int b = bh / H_, h = bh % H_;
  const int w = blockIdx.x * 8 + threadIdx.y;  // 512 waves per bh
  const int wm = w >> 5, wn = w & 31;          // 16 x 32
  const _Float16* A  = q + ((size_t)b * S_ + wm * 64) * D_ + h * DH_;
  const _Float16* Bt = k + ((size_t)b * S_ + wn * 32) * D_ + h * DH_;
  v8f zero = {};
  v8f c[4][2];
#pragma unroll
  for (int i = 0; i < 4; i++)
#pragma unroll
    for (int j = 0; j < 2; j++) c[i][j] = zero;
  wmma_block<4, 2>(A, D_, Bt, D_, DH_, lane, c);
  _Float16* out = att + (size_t)bh * S_ * S_;
  const int hi = lane >> 4;
#pragma unroll
  for (int j = 0; j < 2; j++) {
    const int nn = wn * 32 + j * 16 + (lane & 15);
#pragma unroll
    for (int i = 0; i < 4; i++) {
      const int row = wm * 64 + i * 16 + hi * 8;
#pragma unroll
      for (int r = 0; r < 8; r++)
        out[(size_t)(row + r) * S_ + nn] = (_Float16)(c[i][j][r] * SCALE_);
    }
  }
}

// ctx = probs @ V, residual-added into x (f32). Wave tile 64x32 (N=DH=64),
// K=S=1024.  grid (4, B*H).
__global__ void __launch_bounds__(256, 1)
k_gemm_ctx(const _Float16* __restrict__ att,
           const _Float16* __restrict__ vT,
           float* __restrict__ x) {
  const int lane = threadIdx.x;
  const int bh = blockIdx.y;
  const int b = bh / H_, h = bh % H_;
  const int w = blockIdx.x * 8 + threadIdx.y;   // 32 waves per bh
  const int wm = w >> 1, wn = w & 1;            // 16 x 2
  const _Float16* A  = att + (size_t)bh * S_ * S_ + (size_t)wm * 64 * S_;
  const _Float16* Bt = vT + ((size_t)bh * DH_ + wn * 32) * S_;
  v8f zero = {};
  v8f c[4][2];
#pragma unroll
  for (int i = 0; i < 4; i++)
#pragma unroll
    for (int j = 0; j < 2; j++) c[i][j] = zero;
  wmma_block<4, 2>(A, S_, Bt, S_, S_, lane, c);
  float* out = x + ((size_t)b * S_ + wm * 64) * D_ + h * DH_;
  const int hi = lane >> 4;
#pragma unroll
  for (int j = 0; j < 2; j++) {
    const int nl = wn * 32 + j * 16 + (lane & 15);
#pragma unroll
    for (int i = 0; i < 4; i++)
#pragma unroll
      for (int r = 0; r < 8; r++)
        out[(size_t)(i * 16 + hi * 8 + r) * D_ + nl] += c[i][j][r];
  }
}

// FFN2: x += h16 @ W2 + b2   (M=4096, N=512, K=2048). Wave tile 64x32, grid 128.
__global__ void __launch_bounds__(256, 1)
k_gemm_ffn2(const _Float16* __restrict__ A,
            const _Float16* __restrict__ Bt,
            const float* __restrict__ bias,
            float* __restrict__ x) {
  const int lane = threadIdx.x;
  const int w = blockIdx.x * 8 + threadIdx.y;
  const int wm = w >> 4, wn = w & 15;        // nwn = 512/32 = 16
  v8f zero = {};
  v8f c[4][2];
#pragma unroll
  for (int i = 0; i < 4; i++)
#pragma unroll
    for (int j = 0; j < 2; j++) c[i][j] = zero;
  wmma_block<4, 2>(A + (size_t)wm * 64 * FF_, FF_, Bt + (size_t)wn * 32 * FF_, FF_, FF_, lane, c);
  const int hi = lane >> 4;
#pragma unroll
  for (int j = 0; j < 2; j++) {
    const int nn = wn * 32 + j * 16 + (lane & 15);
    const float bv = bias[nn];
#pragma unroll
    for (int i = 0; i < 4; i++)
#pragma unroll
      for (int r = 0; r < 8; r++)
        x[(size_t)(wm * 64 + i * 16 + hi * 8 + r) * D_ + nn] += c[i][j][r] + bv;
  }
}

// ------------------------------------------------------------------
extern "C" void kernel_launch(void* const* d_in, const int* in_sizes, int n_in,
                              void* d_out, int out_size, void* d_ws, size_t ws_size,
                              hipStream_t stream) {
  (void)in_sizes; (void)n_in; (void)out_size; (void)ws_size;

  const float* src  = (const float*)d_in[0];
  const float* pos  = (const float*)d_in[1];
  const float* Wq   = (const float*)d_in[2];
  const float* bq   = (const float*)d_in[3];
  const float* Wk   = (const float*)d_in[4];
  const float* bk   = (const float*)d_in[5];
  const float* Wv   = (const float*)d_in[6];
  const float* bv   = (const float*)d_in[7];
  const float* W1   = (const float*)d_in[8];
  const float* b1   = (const float*)d_in[9];
  const float* W2   = (const float*)d_in[10];
  const float* b2   = (const float*)d_in[11];
  const float* ln1g = (const float*)d_in[12];
  const float* ln1b = (const float*)d_in[13];
  const float* ln2g = (const float*)d_in[14];
  const float* ln2b = (const float*)d_in[15];
  const float* lnfg = (const float*)d_in[16];
  const float* lnfb = (const float*)d_in[17];
  float* outp = (float*)d_out;

  // ---- workspace carve-up ----
  char* p = (char*)d_ws;
  float*    x   = (float*)p;    p += (size_t)NTOK_ * D_ * 4;           // 8 MB
  _Float16* xn  = (_Float16*)p; p += (size_t)NTOK_ * D_ * 2;           // 4 MB
  _Float16* q16 = (_Float16*)p; p += (size_t)NTOK_ * D_ * 2;           // 4 MB
  _Float16* k16 = (_Float16*)p; p += (size_t)NTOK_ * D_ * 2;           // 4 MB
  _Float16* vT  = (_Float16*)p; p += (size_t)NTOK_ * D_ * 2;           // 4 MB
  _Float16* att = (_Float16*)p; p += (size_t)B_ * H_ * S_ * S_ * 2;    // 64 MB
  _Float16* h16 = (_Float16*)p; p += (size_t)NTOK_ * FF_ * 2;          // 16 MB
  _Float16* wT  = (_Float16*)p;                                        // ~34.6 MB

  const size_t WL = 3 * (size_t)D_ * D_ + 2 * (size_t)D_ * FF_;  // halves/layer
  const dim3 gblk(32, 8);

  // ---- convert + transpose all weights to f16 (N x K) ----
  for (int l = 0; l < L_; l++) {
    _Float16* wq = wT + l * WL;
    _Float16* wk = wq + (size_t)D_ * D_;
    _Float16* wv = wk + (size_t)D_ * D_;
    _Float16* w1 = wv + (size_t)D_ * D_;
    _Float16* w2 = w1 + (size_t)FF_ * D_;
    k_cvt_wt<<<(D_ * D_) / 256, 256, 0, stream>>>(Wq + (size_t)l * D_ * D_, wq, D_, D_);
    k_cvt_wt<<<(D_ * D_) / 256, 256, 0, stream>>>(Wk + (size_t)l * D_ * D_, wk, D_, D_);
    k_cvt_wt<<<(D_ * D_) / 256, 256, 0, stream>>>(Wv + (size_t)l * D_ * D_, wv, D_, D_);
    k_cvt_wt<<<(D_ * FF_) / 256, 256, 0, stream>>>(W1 + (size_t)l * D_ * FF_, w1, D_, FF_);
    k_cvt_wt<<<(FF_ * D_) / 256, 256, 0, stream>>>(W2 + (size_t)l * FF_ * D_, w2, FF_, D_);
  }

  // ---- x = (src + pos) transposed to (B,S,D) ----
  k_add_pos<<<(NTOK_ * D_) / 256, 256, 0, stream>>>(src, pos, x);

  for (int l = 0; l < L_; l++) {
    _Float16* wq = wT + l * WL;
    _Float16* wk = wq + (size_t)D_ * D_;
    _Float16* wv = wk + (size_t)D_ * D_;
    _Float16* w1 = wv + (size_t)D_ * D_;
    _Float16* w2 = w1 + (size_t)FF_ * D_;

    // self-attention block (pre-norm)
    k_ln<<<NTOK_, 256, 0, stream>>>(x, ln1g + l * D_, ln1b + l * D_, xn);
    k_gemm_bias16<false><<<128, gblk, 0, stream>>>(xn, wq, bq + l * D_, q16, D_, D_);
    k_gemm_bias16<false><<<128, gblk, 0, stream>>>(xn, wk, bk + l * D_, k16, D_, D_);
    k_gemm_v<<<128, gblk, 0, stream>>>(xn, wv, bv + l * D_, vT);
    k_gemm_scores<<<dim3(64, B_ * H_), gblk, 0, stream>>>(q16, k16, att);
    k_softmax<<<B_ * H_ * S_, 256, 0, stream>>>(att);
    k_gemm_ctx<<<dim3(4, B_ * H_), gblk, 0, stream>>>(att, vT, x);

    // feed-forward block (pre-norm)
    k_ln<<<NTOK_, 256, 0, stream>>>(x, ln2g + l * D_, ln2b + l * D_, xn);
    k_gemm_bias16<true><<<512, gblk, 0, stream>>>(xn, w1, b1 + l * FF_, h16, FF_, D_);
    k_gemm_ffn2<<<128, gblk, 0, stream>>>(h16, w2, b2 + l * D_, x);
  }

  // final LN + transpose back to (S,B,D)
  k_final_ln<<<NTOK_, 256, 0, stream>>>(x, lnfg, lnfb, outp);
}
